// MultiHeadAttentionCA_21105469293039
// MI455X (gfx1250) — matrix operand order repaired and checked by reading, not compiled, not run on hardware
//
#include <hip/hip_runtime.h>

#define TOK   2048
#define EMB   1024
#define NHEAD 8
#define HE    8192

typedef __attribute__((ext_vector_type(16))) _Float16 v16h;
typedef __attribute__((ext_vector_type(8)))  _Float16 h8;
typedef __attribute__((ext_vector_type(4)))  _Float16 h4;
typedef __attribute__((ext_vector_type(8)))  float    v8f;
typedef __attribute__((ext_vector_type(4)))  float    f4;
typedef __attribute__((ext_vector_type(4)))  int      v4i;

#define BM 128
#define BN 128
#define BK 32
#define PA 40   // LDS pitch in f16 elems (16B-aligned rows)
#define PB 40

// ---- async load-to-LDS (GLOBAL_LOAD_ASYNC_TO_LDS_B128, ASYNCcnt) ----
#if defined(__has_builtin)
#  if __has_builtin(__builtin_amdgcn_global_load_async_to_lds_b128)
#    define HAVE_ASYNC_LDS 1
#  endif
#endif
#ifndef HAVE_ASYNC_LDS
#  define HAVE_ASYNC_LDS 0
#endif

#if HAVE_ASYNC_LDS
typedef __attribute__((address_space(1))) v4i* gv4i_p;   // global int32x4*
typedef __attribute__((address_space(3))) v4i* lv4i_p;   // LDS int32x4*
__device__ __forceinline__ void async_copy16(_Float16* lds, const _Float16* g) {
  __builtin_amdgcn_global_load_async_to_lds_b128((gv4i_p)g, (lv4i_p)lds, 0, 0);
}
#endif

__device__ __forceinline__ void async_wait0() {
#if HAVE_ASYNC_LDS
#  if __has_builtin(__builtin_amdgcn_s_wait_asynccnt)
  __builtin_amdgcn_s_wait_asynccnt(0);
#  else
  asm volatile("s_wait_asynccnt 0x0" ::: "memory");
#  endif
#endif
}

// C[M,Ncols] = (A * B + bias) * alpha
//  A: row-major [M,K]      (f32 -> converted to f16 in LDS, or f16)
//  B: BTRANS ? row-major [Ncols,K] used as B^T : row-major [K,Ncols]
//  batched over blockIdx.z via element strides aBatch/bBatch/cBatch
template<bool AF32, bool BF32, bool BTRANS, bool OUTF32, bool BIAS>
__global__ __launch_bounds__(256)
void gemm_wmma(const void* __restrict__ Av, int lda, long aBatch,
               const void* __restrict__ Bv, int ldb, long bBatch,
               const float* __restrict__ bias, float alpha,
               void* __restrict__ Cv, int ldc, long cBatch, int K)
{
  __shared__ _Float16 As[2][BM * PA];   // As[b][row][k]
  __shared__ _Float16 Bs[2][BN * PB];   // Bs[b][col][k] (k-contiguous per output column)

  const int tid  = threadIdx.x;
  const int lane = tid & 31;
  const int wid  = tid >> 5;
  const int wi   = wid & 1;    // 64-row half of the 128x128 tile
  const int wj   = wid >> 1;   // 32-col quarter
  const int itile = blockIdx.y * BM;
  const int jtile = blockIdx.x * BN;
  const long z    = blockIdx.z;

  const float*    Afp = (const float*)Av    + z * aBatch;
  const _Float16* Ahp = (const _Float16*)Av + z * aBatch;
  const float*    Bfp = (const float*)Bv    + z * bBatch;
  const _Float16* Bhp = (const _Float16*)Bv + z * bBatch;
  float*          Cfp = (float*)Cv          + z * cBatch;
  _Float16*       Chp = (_Float16*)Cv       + z * cBatch;

  // async staging usable only for pure-f16 contiguous copies
  constexpr bool ASYNC_A = (HAVE_ASYNC_LDS != 0) && !AF32;
  constexpr bool ASYNC_B = (HAVE_ASYNC_LDS != 0) && BTRANS;

  v8f acc[4][2];
  #pragma unroll
  for (int mt = 0; mt < 4; ++mt)
    #pragma unroll
    for (int nt = 0; nt < 2; ++nt) {
      v8f zv = {0.f,0.f,0.f,0.f,0.f,0.f,0.f,0.f};
      acc[mt][nt] = zv;
    }

  const int m    = lane & 15;   // M/N index inside a 16x16 WMMA tile
  const int half = lane >> 4;   // lane-half selects K sub-range per ISA layout

  // staging index maps (k0-invariant)
  const int aRow = tid >> 3;            // AF32: 128 rows x 8 quads
  const int aCol = (tid & 7) << 2;
  const int bKK  = tid >> 5;            // NN: 32 k-rows x 32 j-quads
  const int bJ   = (tid & 31) << 2;

  auto stageA = [&](int k0, int b) {
    if constexpr (AF32) {
      // phase 1: issue all global loads (no LDS stores in between -> loads pipeline)
      f4 tmp[4];
      #pragma unroll
      for (int p = 0; p < 4; ++p)
        tmp[p] = *(const f4*)(Afp + (itile + aRow + p * 32) * lda + k0 + aCol);
      // phase 2: convert + store to LDS
      #pragma unroll
      for (int p = 0; p < 4; ++p) {
        h4 hv = { (_Float16)tmp[p][0], (_Float16)tmp[p][1],
                  (_Float16)tmp[p][2], (_Float16)tmp[p][3] };
        *(h4*)&As[b][(aRow + p * 32) * PA + aCol] = hv;
      }
    } else {
#if HAVE_ASYNC_LDS
      #pragma unroll
      for (int p = 0; p < 2; ++p) {
        const int idx = tid + p * 256;     // 512 x 16B cover 128x32 f16
        const int row = idx >> 2;
        const int col = (idx & 3) << 3;
        async_copy16(&As[b][row * PA + col], Ahp + (itile + row) * lda + k0 + col);
      }
#else
      h4 tmp[4];
      #pragma unroll
      for (int p = 0; p < 4; ++p)
        tmp[p] = *(const h4*)(Ahp + (itile + aRow + p * 32) * lda + k0 + aCol);
      #pragma unroll
      for (int p = 0; p < 4; ++p)
        *(h4*)&As[b][(aRow + p * 32) * PA + aCol] = tmp[p];
#endif
    }
  };

  auto stageB = [&](int k0, int b) {
    if constexpr (BTRANS) {
#if HAVE_ASYNC_LDS
      #pragma unroll
      for (int p = 0; p < 2; ++p) {
        const int idx = tid + p * 256;     // 512 x 16B cover 128x32 f16
        const int j  = idx >> 2;
        const int kk = (idx & 3) << 3;
        async_copy16(&Bs[b][j * PB + kk], Bhp + (jtile + j) * ldb + k0 + kk);
      }
#else
      h4 tmp[4];
      #pragma unroll
      for (int p = 0; p < 4; ++p)
        tmp[p] = *(const h4*)(Bhp + (jtile + (tid >> 3) + p * 32) * ldb + k0 + ((tid & 7) << 2));
      #pragma unroll
      for (int p = 0; p < 4; ++p)
        *(h4*)&Bs[b][((tid >> 3) + p * 32) * PB + ((tid & 7) << 2)] = tmp[p];
#endif
    } else if constexpr (BF32) {
      // NN f32: load all k-rows first, then convert + transpose-scatter
      f4 tmp[4];
      #pragma unroll
      for (int p = 0; p < 4; ++p)
        tmp[p] = *(const f4*)(Bfp + (k0 + bKK + p * 8) * ldb + jtile + bJ);
      #pragma unroll
      for (int p = 0; p < 4; ++p) {
        const int kk = bKK + p * 8;
        Bs[b][(bJ+0)*PB + kk] = (_Float16)tmp[p][0];
        Bs[b][(bJ+1)*PB + kk] = (_Float16)tmp[p][1];
        Bs[b][(bJ+2)*PB + kk] = (_Float16)tmp[p][2];
        Bs[b][(bJ+3)*PB + kk] = (_Float16)tmp[p][3];
      }
    } else {
      // NN f16: load all k-rows first, then transpose-scatter
      h4 tmp[4];
      #pragma unroll
      for (int p = 0; p < 4; ++p)
        tmp[p] = *(const h4*)(Bhp + (k0 + bKK + p * 8) * ldb + jtile + bJ);
      #pragma unroll
      for (int p = 0; p < 4; ++p) {
        const int kk = bKK + p * 8;
        Bs[b][(bJ+0)*PB + kk] = tmp[p][0];
        Bs[b][(bJ+1)*PB + kk] = tmp[p][1];
        Bs[b][(bJ+2)*PB + kk] = tmp[p][2];
        Bs[b][(bJ+3)*PB + kk] = tmp[p][3];
      }
    }
  };

  auto fence = [&]() {
    if constexpr (ASYNC_A || ASYNC_B) async_wait0();
    __syncthreads();
  };

  auto compute = [&](int b) {
    v16h afrag[4], bfrag[2];
    #pragma unroll
    for (int mt = 0; mt < 4; ++mt) {
      const int mr = wi * 64 + mt * 16 + m;
      // lanes 0-15: K 0..7 then 16..23 ; lanes 16-31: K 8..15 then 24..31
      h8 lo = *(const h8*)&As[b][mr * PA + half * 8];
      h8 hi = *(const h8*)&As[b][mr * PA + 16 + half * 8];
      v16h a;
      #pragma unroll
      for (int e = 0; e < 8; ++e) { a[e] = lo[e]; a[e + 8] = hi[e]; }
      afrag[mt] = a;
    }
    #pragma unroll
    for (int nt = 0; nt < 2; ++nt) {
      const int nc = wj * 32 + nt * 16 + m;
      // lanes 0-15: K 0..7 then 16..23 ; lanes 16-31: K 8..15 then 24..31
      h8 b0 = *(const h8*)&Bs[b][nc * PB + half * 8];
      h8 b1 = *(const h8*)&Bs[b][nc * PB + 16 + half * 8];
      v16h bv2;
      #pragma unroll
      for (int e = 0; e < 8; ++e) { bv2[e] = b0[e]; bv2[e + 8] = b1[e]; }
      bfrag[nt] = bv2;
    }
    #pragma unroll
    for (int mt = 0; mt < 4; ++mt)
      #pragma unroll
      for (int nt = 0; nt < 2; ++nt)
        acc[mt][nt] = __builtin_amdgcn_wmma_f32_16x16x32_f16(
            false, afrag[mt], false, bfrag[nt], (short)0, acc[mt][nt], false, false);
  };

  // ---- double-buffered main loop: stage k+1 while computing k ----
  stageA(0, 0);
  stageB(0, 0);
  fence();
  int buf = 0;
  for (int k0 = 0; k0 < K; k0 += BK) {
    const int kn = k0 + BK;
    if (kn < K) { stageA(kn, buf ^ 1); stageB(kn, buf ^ 1); }
    compute(buf);
    fence();
    buf ^= 1;
  }

  // ---- epilogue: C/D layout: VGPR r -> M = r + (lane/16)*8, N = lane%16 ----
  #pragma unroll
  for (int mt = 0; mt < 4; ++mt) {
    #pragma unroll
    for (int nt = 0; nt < 2; ++nt) {
      const int jc = jtile + wj * 32 + nt * 16 + m;
      float bb = 0.0f;
      if constexpr (BIAS) bb = bias[jc];
      const int ir0 = itile + wi * 64 + mt * 16 + half * 8;
      #pragma unroll
      for (int r = 0; r < 8; ++r) {
        const float val = (acc[mt][nt][r] + bb) * alpha;
        const int off = (ir0 + r) * ldc + jc;   // < 2^31 for all our shapes
        if constexpr (OUTF32) Cfp[off] = val;
        else                  Chp[off] = (_Float16)val;
      }
    }
  }
}

// In-place softmax over rows of length 2048 (f16 storage, f32 math).
__global__ __launch_bounds__(256)
void softmax2048(_Float16* __restrict__ S)
{
  __shared__ float red[256];
  const long row = blockIdx.x;
  _Float16* p = S + row * 2048;
  const int tid = threadIdx.x;

  h8 v = *(const h8*)(p + tid * 8);
  float f[8];
  float mx = -3.0e38f;
  #pragma unroll
  for (int i = 0; i < 8; ++i) { f[i] = (float)v[i]; mx = fmaxf(mx, f[i]); }

  red[tid] = mx; __syncthreads();
  for (int s = 128; s > 0; s >>= 1) {
    if (tid < s) red[tid] = fmaxf(red[tid], red[tid + s]);
    __syncthreads();
  }
  mx = red[0]; __syncthreads();

  float sum = 0.f;
  #pragma unroll
  for (int i = 0; i < 8; ++i) { f[i] = __expf(f[i] - mx); sum += f[i]; }
  red[tid] = sum; __syncthreads();
  for (int s = 128; s > 0; s >>= 1) {
    if (tid < s) red[tid] += red[tid + s];
    __syncthreads();
  }
  const float inv = 1.0f / red[0];

  h8 o;
  #pragma unroll
  for (int i = 0; i < 8; ++i) o[i] = (_Float16)(f[i] * inv);
  *(h8*)(p + tid * 8) = o;
}

extern "C" void kernel_launch(void* const* d_in, const int* in_sizes, int n_in,
                              void* d_out, int out_size, void* d_ws, size_t ws_size,
                              hipStream_t stream)
{
  const float* x  = (const float*)d_in[0];
  const float* Wq = (const float*)d_in[1];
  const float* bq = (const float*)d_in[2];
  const float* Wk = (const float*)d_in[3];
  const float* bk = (const float*)d_in[4];
  const float* Wv = (const float*)d_in[5];
  const float* bv = (const float*)d_in[6];
  const float* Wo = (const float*)d_in[7];
  const float* bo = (const float*)d_in[8];
  float* out = (float*)d_out;

  // Workspace layout (f16): q,k,v [2048,8192] (32MB each), S [8,2048,2048] (64MB), y [2048,8192] (32MB)
  _Float16* qb = (_Float16*)d_ws;
  _Float16* kb = qb + (size_t)TOK * HE;
  _Float16* vb = kb + (size_t)TOK * HE;
  _Float16* sb = vb + (size_t)TOK * HE;
  _Float16* yb = sb + (size_t)NHEAD * TOK * TOK;

  dim3 blk(256);
  const float scale = 1.0f / 32.0f;   // E^-0.5, folded into Q at projection store

  // 1) Q/K/V projections: x[2048,1024] @ W[1024,8192] (+b) -> f16 [2048,8192]
  dim3 gp(HE / BN, TOK / BM, 1);
  gemm_wmma<true, true, false, false, true><<<gp, blk, 0, stream>>>(
      x, EMB, 0, Wq, HE, 0, bq, scale, qb, HE, 0, EMB);
  gemm_wmma<true, true, false, false, true><<<gp, blk, 0, stream>>>(
      x, EMB, 0, Wk, HE, 0, bk, 1.0f, kb, HE, 0, EMB);
  gemm_wmma<true, true, false, false, true><<<gp, blk, 0, stream>>>(
      x, EMB, 0, Wv, HE, 0, bv, 1.0f, vb, HE, 0, EMB);

  // 2) Scores per head: S[h] = Q[h] @ K[h]^T  (batched over z)
  dim3 gs(TOK / BN, TOK / BM, NHEAD);
  gemm_wmma<false, false, true, false, false><<<gs, blk, 0, stream>>>(
      qb, HE, EMB, kb, HE, EMB, nullptr, 1.0f, sb, TOK, (long)TOK * TOK, EMB);

  // 3) Row softmax over all H*N rows, in place
  softmax2048<<<dim3(NHEAD * TOK), blk, 0, stream>>>(sb);

  // 4) O[h] = P[h] @ V[h] -> concat layout y[2048, 8192]
  dim3 go(EMB / BN, TOK / BM, NHEAD);
  gemm_wmma<false, false, false, false, false><<<go, blk, 0, stream>>>(
      sb, TOK, (long)TOK * TOK, vb, HE, EMB, nullptr, 1.0f, yb, HE, EMB, TOK);

  // 5) out = y @ Wo + bo -> f32 [2048,1024]
  dim3 gf(EMB / BN, TOK / BM, 1);
  gemm_wmma<false, true, false, true, true><<<gf, blk, 0, stream>>>(
      yb, HE, 0, Wo, EMB, 0, bo, 1.0f, out, EMB, 0, HE);
}
